// RNNStepPredictor_1717986918784
// MI455X (gfx1250) — matrix operand order repaired and checked by reading, not compiled, not run on hardware
//
#include <hip/hip_runtime.h>
#include <hip/hip_bf16.h>

// Problem constants (match reference)
#define BB 512
#define TT 64
#define HH 1024
#define FF 2
#define GH (4 * HH)   // 4096

typedef __attribute__((ext_vector_type(16))) _Float16 v16h;
typedef __attribute__((ext_vector_type(8)))  _Float16 v8h;
typedef __attribute__((ext_vector_type(8)))  float    v8f;

union ABfrag { v16h v; v8h h[2]; };

// ---------------------------------------------------------------------------
// One-time: transpose+convert rec_kernel [H, 4H] f32  ->  recT [4H, H] f16
// LDS-tiled 32x32 transpose, coalesced reads and writes.
// ---------------------------------------------------------------------------
__global__ __launch_bounds__(256)
void transpose_convert_f16(const float* __restrict__ rec, _Float16* __restrict__ recT) {
  __shared__ _Float16 tile[32][33];
  const int tx = threadIdx.x;        // 0..31
  const int ty = threadIdx.y;        // 0..7
  const int c0 = blockIdx.x * 32;    // column tile in [0, 4H)
  const int k0 = blockIdx.y * 32;    // row    tile in [0, H)
#pragma unroll
  for (int i = 0; i < 32; i += 8) {
    // rec[k][c], k = k0+ty+i, c = c0+tx  (coalesced over tx)
    tile[ty + i][tx] = (_Float16)rec[(size_t)(k0 + ty + i) * GH + (c0 + tx)];
  }
  __syncthreads();
#pragma unroll
  for (int i = 0; i < 32; i += 8) {
    // recT[c][k], c = c0+ty+i, k = k0+tx  (coalesced over tx)
    recT[(size_t)(c0 + ty + i) * HH + (k0 + tx)] = tile[tx][ty + i];
  }
}

// ---------------------------------------------------------------------------
// One LSTM timestep, barrier-free.
// Each WAVE computes ALL FOUR gate tiles (i,f,g,o) for one 16x16 (batch x
// hidden) tile: the A (h) fragment is loaded once per K-slab and reused by
// 4 WMMAs; the cell update happens entirely in registers (the C/D layout
// maps the same (m,n) to the same (lane,r) in all four accumulators).
// Block: 256 threads = 8 waves, wave wid owns hidden tile j0 = (bx*8+wid)*16.
// grid = (H/16/8, B/16) = (8, 32)  -> 2048 independent waves per step.
// ---------------------------------------------------------------------------
__global__ __launch_bounds__(256)
void lstm_step(const float*    __restrict__ x,        // [B,T,F]
               const float*    __restrict__ Wx,       // [F,4H]
               const float*    __restrict__ bias,     // [4H]
               const _Float16* __restrict__ recT,     // [4H,H] f16 (transposed)
               const _Float16* __restrict__ hin,      // [B,H] f16 (h_{t-1})
               float*          __restrict__ cbuf,     // [B,H] f32 cell state (in/out)
               _Float16*       __restrict__ hout,     // [B,H] f16 (h_t for next step)
               float*          __restrict__ lstm_out, // [B,T,H]
               float*          __restrict__ hfin,     // [B,H] (overwritten each step)
               int t) {
  const int tid  = threadIdx.x;
  const int wid  = tid >> 5;
  const int lane = tid & 31;
  const int j0   = (blockIdx.x * 8 + wid) * 16;  // hidden-column tile
  const int b0   = blockIdx.y * 16;              // batch-row tile

  const int nm = lane & 15;           // M (for A) / N (for B)
  const int hi = lane >> 4;           // K-group selector per ISA layout

  const _Float16* arow = hin + (size_t)(b0 + nm) * HH;            // A: h row
  // B rows for the four gates: recT rows  g*H + j0 + nm
  const _Float16* brow0 = recT + (size_t)(0 * HH + j0 + nm) * HH;
  const _Float16* brow1 = recT + (size_t)(1 * HH + j0 + nm) * HH;
  const _Float16* brow2 = recT + (size_t)(2 * HH + j0 + nm) * HH;
  const _Float16* brow3 = recT + (size_t)(3 * HH + j0 + nm) * HH;

  v8f acc0 = {}, acc1 = {}, acc2 = {}, acc3 = {};
#pragma unroll 2
  for (int k0 = 0; k0 < HH; k0 += 32) {
    ABfrag a, b0f, b1f, b2f, b3f;
    // A 16x32 f16: lane<16 holds K0..15; lane>=16 holds K16..31 (contiguous)
    a.h[0] = *(const v8h*)(arow + k0 + hi * 16);
    a.h[1] = *(const v8h*)(arow + k0 + hi * 16 + 8);
    // B 32x16 f16: lane<16 holds K0..15; lane>=16 holds K16..31 (contiguous)
    const int bo = k0 + hi * 16;
    b0f.h[0] = *(const v8h*)(brow0 + bo);
    b0f.h[1] = *(const v8h*)(brow0 + bo + 8);
    b1f.h[0] = *(const v8h*)(brow1 + bo);
    b1f.h[1] = *(const v8h*)(brow1 + bo + 8);
    b2f.h[0] = *(const v8h*)(brow2 + bo);
    b2f.h[1] = *(const v8h*)(brow2 + bo + 8);
    b3f.h[0] = *(const v8h*)(brow3 + bo);
    b3f.h[1] = *(const v8h*)(brow3 + bo + 8);
    acc0 = __builtin_amdgcn_wmma_f32_16x16x32_f16(false, a.v, false, b0f.v,
                                                  (short)0, acc0, false, false);
    acc1 = __builtin_amdgcn_wmma_f32_16x16x32_f16(false, a.v, false, b1f.v,
                                                  (short)0, acc1, false, false);
    acc2 = __builtin_amdgcn_wmma_f32_16x16x32_f16(false, a.v, false, b2f.v,
                                                  (short)0, acc2, false, false);
    acc3 = __builtin_amdgcn_wmma_f32_16x16x32_f16(false, a.v, false, b3f.v,
                                                  (short)0, acc3, false, false);
  }

  // Per-gate input-projection coefficients for this lane's column (F=2).
  const float kc00 = Wx[0 * HH + j0 + nm], kc01 = Wx[GH + 0 * HH + j0 + nm];
  const float kc10 = Wx[1 * HH + j0 + nm], kc11 = Wx[GH + 1 * HH + j0 + nm];
  const float kc20 = Wx[2 * HH + j0 + nm], kc21 = Wx[GH + 2 * HH + j0 + nm];
  const float kc30 = Wx[3 * HH + j0 + nm], kc31 = Wx[GH + 3 * HH + j0 + nm];
  const float bc0 = bias[0 * HH + j0 + nm];
  const float bc1 = bias[1 * HH + j0 + nm];
  const float bc2 = bias[2 * HH + j0 + nm];
  const float bc3 = bias[3 * HH + j0 + nm];

  // In-register epilogue. C/D layout: lane<16 -> N=lane, M=r; lane>=16 -> M=r+8.
#pragma unroll
  for (int r = 0; r < 8; ++r) {
    const int m = r + hi * 8;
    const int b = b0 + m;
    const float x0 = x[(size_t)b * TT * FF + t * FF + 0];
    const float x1 = x[(size_t)b * TT * FF + t * FF + 1];
    const float zi = acc0[r] + x0 * kc00 + x1 * kc01 + bc0;
    const float zf = acc1[r] + x0 * kc10 + x1 * kc11 + bc1;
    const float zg = acc2[r] + x0 * kc20 + x1 * kc21 + bc2;
    const float zo = acc3[r] + x0 * kc30 + x1 * kc31 + bc3;
    const float ig = 1.0f / (1.0f + __expf(-zi));
    const float fg = 1.0f / (1.0f + __expf(-zf));
    const float og = 1.0f / (1.0f + __expf(-zo));
    const int j = j0 + nm;
    const size_t cidx = (size_t)b * HH + j;
    const float cn = fg * cbuf[cidx] + ig * tanhf(zg);
    const float h  = og * tanhf(cn);
    cbuf[cidx] = cn;
    lstm_out[((size_t)b * TT + t) * HH + j] = h;
    hfin[cidx] = h;
    hout[cidx] = (_Float16)h;
  }
}

// ---------------------------------------------------------------------------
// pred[b,t,:] = tanh(h[b,:] @ dense_W[t] + dense_b[t]);  one block per batch row
// ---------------------------------------------------------------------------
__global__ __launch_bounds__(256)
void pred_step(const float* __restrict__ lstm_out,  // [B,T,H]
               const float* __restrict__ dW,        // [T,H,2]
               const float* __restrict__ db,        // [T,2]
               float*       __restrict__ preds,     // [B,T,2]
               int t) {
  const int b   = blockIdx.x;
  const int tid = threadIdx.x;
  const float* h = lstm_out + ((size_t)b * TT + t) * HH;
  const float* w = dW + (size_t)t * HH * 2;
  float s0 = 0.0f, s1 = 0.0f;
  for (int j = tid; j < HH; j += 256) {
    const float hv = h[j];
    s0 += hv * w[j * 2 + 0];
    s1 += hv * w[j * 2 + 1];
  }
  __shared__ float r0[256], r1[256];
  r0[tid] = s0; r1[tid] = s1;
  __syncthreads();
  for (int off = 128; off > 0; off >>= 1) {
    if (tid < off) { r0[tid] += r0[tid + off]; r1[tid] += r1[tid + off]; }
    __syncthreads();
  }
  if (tid == 0) {
    preds[((size_t)b * TT + t) * 2 + 0] = tanhf(r0[0] + db[t * 2 + 0]);
    preds[((size_t)b * TT + t) * 2 + 1] = tanhf(r1[0] + db[t * 2 + 1]);
  }
}

// ---------------------------------------------------------------------------
extern "C" void kernel_launch(void* const* d_in, const int* in_sizes, int n_in,
                              void* d_out, int out_size, void* d_ws, size_t ws_size,
                              hipStream_t stream) {
  (void)in_sizes; (void)n_in; (void)out_size; (void)ws_size;

  const float* x    = (const float*)d_in[0];   // [B,T,F]
  const float* Wx   = (const float*)d_in[1];   // [F,4H]
  const float* Wh   = (const float*)d_in[2];   // [H,4H]
  const float* bias = (const float*)d_in[3];   // [4H]
  const float* dW   = (const float*)d_in[4];   // [T,H,2]
  const float* db   = (const float*)d_in[5];   // [T,2]

  float* out   = (float*)d_out;
  float* preds = out;                                   // [B,T,2]
  float* lstm  = out + (size_t)BB * TT * 2;             // [B,T,H]
  float* hfin  = lstm + (size_t)BB * TT * HH;           // [B,H]
  float* cfin  = hfin + (size_t)BB * HH;                // [B,H]

  char* wsb = (char*)d_ws;
  _Float16* recT = (_Float16*)wsb;                          // 4H*H f16 = 8 MB
  _Float16* hA   = (_Float16*)(wsb + (size_t)8 * 1024 * 1024);  // 1 MB
  _Float16* hB   = (_Float16*)(wsb + (size_t)9 * 1024 * 1024);  // 1 MB

  // One-time weight transpose+convert (weights then live in L2 for all steps).
  {
    dim3 tb(32, 8);
    dim3 tg(GH / 32, HH / 32);
    transpose_convert_f16<<<tg, tb, 0, stream>>>(Wh, recT);
  }
  // h0 = 0 (f16 zeros == all-zero bytes), c0 = 0.
  hipMemsetAsync(hA, 0, (size_t)BB * HH * sizeof(_Float16), stream);
  hipMemsetAsync(cfin, 0, (size_t)BB * HH * sizeof(float), stream);

  _Float16* hin  = hA;
  _Float16* hout = hB;
  const dim3 grid(HH / 16 / 8, BB / 16);  // (8, 32)
  for (int t = 0; t < TT; ++t) {
    lstm_step<<<grid, 256, 0, stream>>>(x, Wx, bias, recT, hin, cfin, hout,
                                        lstm, hfin, t);
    pred_step<<<BB, 256, 0, stream>>>(lstm, dW, db, preds, t);
    _Float16* tmp = hin; hin = hout; hout = tmp;
  }
}
